// MH_gate_7533372637663
// MI455X (gfx1250) — matrix operand order repaired and checked by reading, not compiled, not run on hardware
//
#include <hip/hip_runtime.h>
#include <hip/hip_bf16.h>

// ---------------------------------------------------------------------------
// MH_gate fused graph-attention block for MI455X (gfx1250, wave32, WMMA+TDM)
// B=16, N=512, D=512, H=8, DK=64
//  - weights pre-packed in global into WMMA B-fragment order (once per call)
//  - GEMM hot loop: TDM tensor_load_to_lds (double buffered) + wmma
//  - every WMMA operand load is 2x ds_load_b128
// ---------------------------------------------------------------------------

#define B_ 16
#define N_ 512
#define D_ 512
#define H_ 8
#define DK_ 64
#define SCALE_ 0.125f
#define LN_EPS_ 1e-5f

typedef __attribute__((ext_vector_type(16))) __bf16 v16bf;
typedef __attribute__((ext_vector_type(8)))  float  v8f;
typedef unsigned v4u __attribute__((ext_vector_type(4)));
typedef int      v8i __attribute__((ext_vector_type(8)));
typedef int      v4i __attribute__((ext_vector_type(4)));

union FragAB { v16bf v; unsigned short u[16]; uint4 q[2]; };
union FragC  { v8f   v; float f[8]; };
union Pack16 { unsigned short u[16]; uint4 q[2]; };

__device__ __forceinline__ unsigned short f2bf(float f) {
    unsigned u = __float_as_uint(f);
    u += 0x7FFFu + ((u >> 16) & 1u);       // round-to-nearest-even
    return (unsigned short)(u >> 16);
}

// ---------------------------------------------------------------------------
// TDM: 1-row tensor copy of n8*8 bytes global->LDS (D# per ISA 08 §8)
// ---------------------------------------------------------------------------
__device__ __forceinline__ void tdm_load_1d(unsigned lds_off, const void* gptr,
                                            unsigned n8) {
    unsigned long long ga = (unsigned long long)gptr;
    v4u g0;
    g0[0] = 1u;                                     // count=1 (valid), user mode
    g0[1] = lds_off;                                // lds_addr (bytes)
    g0[2] = (unsigned)(ga & 0xFFFFFFFFu);           // global_addr[31:0]
    g0[3] = (unsigned)((ga >> 32) & 0x1FFFFFFu) | (2u << 30);  // addr[56:32] | type=2
    v8i g1;
    g1[0] = (3 << 16);                              // data_size = 8 bytes
    g1[1] = (int)((n8 & 0xFFFFu) << 16);            // tensor_dim0[15:0]
    g1[2] = (int)(((n8 >> 16) & 0xFFFFu) | (1u << 16)); // dim0[31:16] | tensor_dim1=1
    g1[3] = (int)((n8 & 0xFFFFu) << 16);            // tile_dim0 = n8
    g1[4] = 1;                                      // tile_dim1 = 1
    g1[5] = (int)n8;                                // tensor_dim0_stride
    g1[6] = 0;
    g1[7] = 0;
    v4i z = {0, 0, 0, 0};
#if defined(__clang_major__) && (__clang_major__ >= 23)
    v8i z8 = {0, 0, 0, 0, 0, 0, 0, 0};
    __builtin_amdgcn_tensor_load_to_lds(g0, g1, z, z, z8, 0);
#else
    __builtin_amdgcn_tensor_load_to_lds(g0, g1, z, z, 0);
#endif
}

__device__ __forceinline__ void tdm_wait(bool last) {
    if (last) __builtin_amdgcn_s_wait_tensorcnt(0);
    else      __builtin_amdgcn_s_wait_tensorcnt(1);
}

__device__ __forceinline__ unsigned lds_off_of(const void* p) {
    return (unsigned)(unsigned long long)p;
}

// ---------------------------------------------------------------------------
// Kernel: fp32 -> bf16 conversion (activations)
// ---------------------------------------------------------------------------
__global__ void k_cvt_bf16(const float* __restrict__ src,
                           unsigned short* __restrict__ dst, int n) {
    int i = blockIdx.x * blockDim.x + threadIdx.x;
    if (i < n) dst[i] = f2bf(src[i]);
}

// ---------------------------------------------------------------------------
// Kernel: pack a 512x512 fp32 weight into bf16 WMMA-B fragment order.
// layout: wpk[((kc*64 + ct*2 + kkh)*32 + l)*16 + e] =
//         W[(kc*64 + kkh*32 + (l>>4)*16 + e)*512 + ct*16 + (l&15)]
// 16384 lane-slots; consecutive threads touch consecutive n (coalesced).
// ---------------------------------------------------------------------------
__global__ void k_pack_w(const float* __restrict__ W,
                         unsigned short* __restrict__ wpk) {
    int gs = blockIdx.x * blockDim.x + threadIdx.x;   // 0..16383
    int kc = gs >> 11, s = gs & 2047;
    int n = s & 511, half = (s >> 9) & 1, kkh = s >> 10;
    int kb = kc * 64 + kkh * 32 + half * 16;
    Pack16 tmp;
#pragma unroll
    for (int e = 0; e < 16; ++e)
        tmp.u[e] = f2bf(W[(size_t)(kb + e) * D_ + n]);
    int ct = n >> 4, l = (n & 15) + half * 16;
    unsigned short* dst = &wpk[((size_t)((kc * 64 + ct * 2 + kkh) * 32 + l)) * 16];
    *(uint4*)dst       = tmp.q[0];
    *(uint4*)(dst + 8) = tmp.q[1];
}

// ---------------------------------------------------------------------------
// Fragment loads (ISA 7.12.2, wave32): 2x ds_load_b128 each
// ---------------------------------------------------------------------------
__device__ __forceinline__ void load_fragA(FragAB& a, const unsigned short* s,
                                           int rowStride, int kbase, int lane) {
    const unsigned short* p = s + (lane & 15) * rowStride + kbase + ((lane >> 4) & 1) * 16;
    a.q[0] = *(const uint4*)p;
    a.q[1] = *(const uint4*)(p + 8);
}
__device__ __forceinline__ void load_fragB_packed(FragAB& b, const unsigned short* sBp,
                                                  int frag, int lane) {
    const unsigned short* p = sBp + ((size_t)(frag * 32 + lane)) * 16;
    b.q[0] = *(const uint4*)p;
    b.q[1] = *(const uint4*)(p + 8);
}

// ---------------------------------------------------------------------------
// Shared GEMM core: 16x512 tile, K-chunk 64, TDM double-buffered packed B.
// wpk: packed weight (8 chunks x 64KB). sB0/sB1: 64KB each.
// ---------------------------------------------------------------------------
__device__ __forceinline__ void gemm_tile(const unsigned short* __restrict__ abf,
                                          const unsigned short* __restrict__ wpk,
                                          unsigned short* sA,
                                          unsigned short* sB0, unsigned short* sB1,
                                          FragC acc[4], int row0, int t, int lane, int wave) {
    const unsigned sb_off[2] = { lds_off_of(sB0), lds_off_of(sB1) };

    { // stage A chunk 0
        int e = t * 4, r = e >> 6, kk = e & 63;
        *(uint2*)&sA[r * 64 + kk] =
            *(const uint2*)&abf[(size_t)(row0 + r) * D_ + kk];
    }
    if (wave == 0) tdm_load_1d(sb_off[0], wpk, 8192);          // chunk 0

    for (int i = 0; i < 8; ++i) {
        if (wave == 0) {
            if (i + 1 < 8)
                tdm_load_1d(sb_off[(i + 1) & 1], wpk + (size_t)(i + 1) * 32768, 8192);
            tdm_wait(i + 1 >= 8);
        }
        __syncthreads();                                       // A(i), B(i) ready

        const unsigned short* sB = (i & 1) ? sB1 : sB0;
#pragma unroll
        for (int ctl = 0; ctl < 4; ++ctl) {
            int ct = wave * 4 + ctl;
#pragma unroll
            for (int kkh = 0; kkh < 2; ++kkh) {
                FragAB a, b;
                load_fragA(a, sA, 64, kkh * 32, lane);
                load_fragB_packed(b, sB, ct * 2 + kkh, lane);
                acc[ctl].v = __builtin_amdgcn_wmma_f32_16x16x32_bf16(
                    false, a.v, false, b.v, (short)0, acc[ctl].v, false, false);
            }
        }
        __syncthreads();                                       // done reading sA
        if (i + 1 < 8) { // stage A chunk i+1
            int e = t * 4, r = e >> 6, kk = e & 63;
            *(uint2*)&sA[r * 64 + kk] =
                *(const uint2*)&abf[(size_t)(row0 + r) * D_ + (i + 1) * 64 + kk];
        }
    }
}

// dyn LDS (GEMM kernels): sA 2K | sB0 64K (reused as sC 32K) | sB1 64K | red 2K | stat
#define GEMM_SMEM 135296
#define QKV_SMEM  133120

// ---------------------------------------------------------------------------
// Kernel: h = LayerNorm(x @ W_w + b)
// ---------------------------------------------------------------------------
__global__ __launch_bounds__(256, 1)
void k_gemm_ln(const unsigned short* __restrict__ xbf,
               const unsigned short* __restrict__ wpk,
               const float* __restrict__ bias,
               const float* __restrict__ g,
               const float* __restrict__ bta,
               unsigned short* __restrict__ hbf) {
    extern __shared__ unsigned char smem[];
    unsigned short* sA  = (unsigned short*)smem;
    unsigned short* sB0 = (unsigned short*)(smem + 2048);
    unsigned short* sB1 = (unsigned short*)(smem + 2048 + 65536);
    float*          sC  = (float*)(smem + 2048);               // overlay sB0
    float*          red = (float*)(smem + 2048 + 131072);
    float*          stat= red + 512;

    const int t = threadIdx.x, lane = t & 31, wave = t >> 5;
    const int row0 = blockIdx.x * 16;

    FragC acc[4];
#pragma unroll
    for (int ct = 0; ct < 4; ++ct)
#pragma unroll
        for (int r = 0; r < 8; ++r) acc[ct].f[r] = 0.f;

    gemm_tile(xbf, wpk, sA, sB0, sB1, acc, row0, t, lane, wave);

#pragma unroll
    for (int ct = 0; ct < 4; ++ct) {
        int col = wave * 64 + ct * 16 + (lane & 15);
        float bv = bias[col];
#pragma unroll
        for (int r = 0; r < 8; ++r) {
            int row = ((lane >> 4) & 1) * 8 + r;
            sC[row * 512 + col] = acc[ct].f[r] + bv;
        }
    }
    __syncthreads();

    int lrow = t >> 4, lcol = t & 15;
    float s = 0.f, s2 = 0.f;
    for (int c = lcol; c < 512; c += 16) {
        float v = sC[lrow * 512 + c];
        s += v; s2 += v * v;
    }
    red[lrow * 16 + lcol] = s;
    red[256 + lrow * 16 + lcol] = s2;
    __syncthreads();
    if (lcol == 0) {
        float m = 0.f, q = 0.f;
        for (int i = 0; i < 16; ++i) { m += red[lrow * 16 + i]; q += red[256 + lrow * 16 + i]; }
        m *= (1.f / 512.f);
        q = q * (1.f / 512.f) - m * m;
        stat[lrow] = m;
        stat[16 + lrow] = rsqrtf(q + LN_EPS_);
    }
    __syncthreads();
    float mu = stat[lrow], rs = stat[16 + lrow];
    for (int c = lcol; c < 512; c += 16) {
        float v = (sC[lrow * 512 + c] - mu) * rs * g[c] + bta[c];
        hbf[(size_t)(row0 + lrow) * D_ + c] = f2bf(v);
    }
}

// ---------------------------------------------------------------------------
// Kernel: q/k = (h@W+b)*scale -> [B,H,N,DK]; v -> fragment-packed per head
// ---------------------------------------------------------------------------
__global__ __launch_bounds__(256, 1)
void k_gemm_qkv(const unsigned short* __restrict__ hbf,
                const unsigned short* __restrict__ wpk,
                const float* __restrict__ bias, float scale, int packed_v,
                unsigned short* __restrict__ out) {
    extern __shared__ unsigned char smem[];
    unsigned short* sA  = (unsigned short*)smem;
    unsigned short* sB0 = (unsigned short*)(smem + 2048);
    unsigned short* sB1 = (unsigned short*)(smem + 2048 + 65536);

    const int t = threadIdx.x, lane = t & 31, wave = t >> 5;
    const int row0 = blockIdx.x * 16;

    FragC acc[4];
#pragma unroll
    for (int ct = 0; ct < 4; ++ct)
#pragma unroll
        for (int r = 0; r < 8; ++r) acc[ct].f[r] = 0.f;

    gemm_tile(hbf, wpk, sA, sB0, sB1, acc, row0, t, lane, wave);

#pragma unroll
    for (int ct = 0; ct < 4; ++ct) {
        int col = wave * 64 + ct * 16 + (lane & 15);
        int head = col >> 6, dk = col & 63;
        float bv = bias[col];
#pragma unroll
        for (int r = 0; r < 8; ++r) {
            int row = ((lane >> 4) & 1) * 8 + r;
            int brow = row0 + row;
            int b = brow >> 9, n = brow & 511;
            float v = (acc[ct].f[r] + bv) * scale;
            if (!packed_v) {
                out[(((size_t)b * H_ + head) * N_ + n) * DK_ + dk] = f2bf(v);
            } else {
                // packed-V: frag = dkt*16 + k0c; lane = (dk&15)+half*16; elem = n&15
                int e_i = n & 15, half = (n >> 4) & 1, k0c = n >> 5;
                int l = (dk & 15) + half * 16, dkt = dk >> 4;
                size_t bh = (size_t)b * H_ + head;
                out[((bh * 64 + dkt * 16 + k0c) * 32 + l) * 16 + e_i] = f2bf(v);
            }
        }
    }
}

// ---------------------------------------------------------------------------
// Kernel: attention per (b,h,16-query tile).
// dyn LDS: sQ 2K | sK0 16K | sK1 16K | sS 32K | sP 16K | sVp 64K | red 2K | stat
// K chunks (contiguous 16KB) + packed V head (contiguous 64KB) via TDM.
// ---------------------------------------------------------------------------
#define ATT_SMEM 151680
__global__ __launch_bounds__(256, 1)
void k_attn(const unsigned short* __restrict__ qbf,
            const unsigned short* __restrict__ kbf,
            const unsigned short* __restrict__ vpk,
            const float* __restrict__ adj,
            const int* __restrict__ use_adj,
            unsigned short* __restrict__ ctxbf) {
    extern __shared__ unsigned char smem[];
    unsigned short* sQ  = (unsigned short*)(smem + 0);        // 16x64
    unsigned short* sK0 = (unsigned short*)(smem + 2048);     // 128x64
    unsigned short* sK1 = (unsigned short*)(smem + 18432);    // 128x64
    float*          sS  = (float*)(smem + 34816);             // 16x512
    unsigned short* sP  = (unsigned short*)(smem + 67584);    // 16x512
    unsigned short* sVp = (unsigned short*)(smem + 83968);    // packed 64KB
    float*          red = (float*)(smem + 149504);            // 512
    float*          stat= (float*)(smem + 151552);            // 32

    const int t = threadIdx.x, lane = t & 31, wave = t >> 5;
    const int qt = blockIdx.x & 31;
    const int bh = blockIdx.x >> 5;
    const int b  = bh >> 3, hh = bh & 7;
    const int qrow0 = qt * 16;
    const size_t headbase = (size_t)bh * N_ * DK_;
    const int mask_on = use_adj[0];
    const unsigned sk_off[2] = { lds_off_of(sK0), lds_off_of(sK1) };

    { // stage Q tile 16x64 (tiny, synchronous)
        int e = t * 4, r = e >> 6, dk = e & 63;
        *(uint2*)&sQ[r * 64 + dk] =
            *(const uint2*)&qbf[headbase + (size_t)(qrow0 + r) * DK_ + dk];
    }
    if (wave == 0) {
        tdm_load_1d(lds_off_of(sVp), vpk + (size_t)bh * 32768, 8192);  // V head 64KB
        tdm_load_1d(sk_off[0], kbf + headbase, 2048);                  // K chunk 0 16KB
    }

    // ---- scores: q @ k^T, masked (ref: where(a>0,s,NEG)*a == a>0 ? s : 0) ----
    for (int kci = 0; kci < 4; ++kci) {
        if (wave == 0) {
            if (kci + 1 < 4)
                tdm_load_1d(sk_off[(kci + 1) & 1],
                            kbf + headbase + (size_t)(kci + 1) * 8192, 2048);
            tdm_wait(kci + 1 >= 4);
        }
        __syncthreads();

        const unsigned short* sK = (kci & 1) ? sK1 : sK0;
        FragC sc;
#pragma unroll
        for (int r = 0; r < 8; ++r) sc.f[r] = 0.f;
        int keyl = wave * 16 + (lane & 15);
#pragma unroll
        for (int kk = 0; kk < DK_; kk += 32) {
            FragAB a, bfr;
            load_fragA(a, sQ, 64, kk, lane);
            const unsigned short* p = &sK[keyl * 64 + kk + ((lane >> 4) & 1) * 16];
            bfr.q[0] = *(const uint4*)p;
            bfr.q[1] = *(const uint4*)(p + 8);
            sc.v = __builtin_amdgcn_wmma_f32_16x16x32_bf16(
                false, a.v, false, bfr.v, (short)0, sc.v, false, false);
        }
        int keyg = kci * 128 + keyl;
#pragma unroll
        for (int r = 0; r < 8; ++r) {
            int row = ((lane >> 4) & 1) * 8 + r;
            float s = sc.f[r];
            if (mask_on) {
                float a = adj[((size_t)b * N_ + qrow0 + row) * N_ + keyg];
                s = (a > 0.f) ? s : 0.f;
            }
            sS[row * 512 + keyg] = s;
        }
        __syncthreads();
    }

    // ---- softmax over 512 keys, 16 threads/row ----
    int lrow = t >> 4, lcol = t & 15;
    float mx = -3.0e38f;
    for (int c = lcol; c < 512; c += 16) mx = fmaxf(mx, sS[lrow * 512 + c]);
    red[lrow * 16 + lcol] = mx;
    __syncthreads();
    if (lcol == 0) {
        float m = red[lrow * 16];
        for (int i = 1; i < 16; ++i) m = fmaxf(m, red[lrow * 16 + i]);
        stat[lrow] = m;
    }
    __syncthreads();
    float m = stat[lrow], sum = 0.f;
    for (int c = lcol; c < 512; c += 16) {
        float e = __expf(sS[lrow * 512 + c] - m);
        sS[lrow * 512 + c] = e;
        sum += e;
    }
    red[256 + lrow * 16 + lcol] = sum;
    __syncthreads();
    if (lcol == 0) {
        float tot = 0.f;
        for (int i = 0; i < 16; ++i) tot += red[256 + lrow * 16 + i];
        stat[16 + lrow] = 1.f / tot;
    }
    __syncthreads();
    float inv = stat[16 + lrow];
    for (int c = lcol; c < 512; c += 16)
        sP[lrow * 512 + c] = f2bf(sS[lrow * 512 + c] * inv);
    __syncthreads();

    // ---- ctx = attn @ v : waves 0..3 each own a 16-dk column tile ----
    if (wave < 4) {
        FragC o;
#pragma unroll
        for (int r = 0; r < 8; ++r) o.f[r] = 0.f;
#pragma unroll
        for (int k0c = 0; k0c < 16; ++k0c) {
            FragAB a, bfr;
            load_fragA(a, sP, 512, k0c * 32, lane);
            load_fragB_packed(bfr, sVp, wave * 16 + k0c, lane);
            o.v = __builtin_amdgcn_wmma_f32_16x16x32_bf16(
                false, a.v, false, bfr.v, (short)0, o.v, false, false);
        }
        int col = wave * 16 + (lane & 15);
#pragma unroll
        for (int r = 0; r < 8; ++r) {
            int row = ((lane >> 4) & 1) * 8 + r;
            ctxbf[((size_t)b * N_ + qrow0 + row) * D_ + hh * DK_ + col] = f2bf(o.f[r]);
        }
    }
}

// ---------------------------------------------------------------------------
// Kernel: h2 = relu(ctx@Wo+bo); coeff = sigmoid([x,h2]@gw+gb); blend
// ---------------------------------------------------------------------------
__global__ __launch_bounds__(256, 1)
void k_out(const unsigned short* __restrict__ ctxbf,
           const unsigned short* __restrict__ wpk,
           const float* __restrict__ bo,
           const float* __restrict__ x,
           const float* __restrict__ gw,
           const float* __restrict__ gb,
           float* __restrict__ out) {
    extern __shared__ unsigned char smem[];
    unsigned short* sA  = (unsigned short*)smem;
    unsigned short* sB0 = (unsigned short*)(smem + 2048);
    unsigned short* sB1 = (unsigned short*)(smem + 2048 + 65536);
    float*          sC  = (float*)(smem + 2048);               // overlay sB0
    float*          red = (float*)(smem + 2048 + 131072);
    float*          stat= red + 512;

    const int t = threadIdx.x, lane = t & 31, wave = t >> 5;
    const int row0 = blockIdx.x * 16;

    FragC acc[4];
#pragma unroll
    for (int ct = 0; ct < 4; ++ct)
#pragma unroll
        for (int r = 0; r < 8; ++r) acc[ct].f[r] = 0.f;

    gemm_tile(ctxbf, wpk, sA, sB0, sB1, acc, row0, t, lane, wave);

#pragma unroll
    for (int ct = 0; ct < 4; ++ct) {
        int col = wave * 64 + ct * 16 + (lane & 15);
        float bv = bo[col];
#pragma unroll
        for (int r = 0; r < 8; ++r) {
            int row = ((lane >> 4) & 1) * 8 + r;
            sC[row * 512 + col] = fmaxf(acc[ct].f[r] + bv, 0.f);
        }
    }
    __syncthreads();

    int lrow = t >> 4, lcol = t & 15;
    size_t grow = (size_t)(row0 + lrow);
    float gx = 0.f, gh = 0.f;
    for (int c = lcol; c < 512; c += 16) {
        gx += x[grow * D_ + c] * gw[c];
        gh += sC[lrow * 512 + c] * gw[512 + c];
    }
    red[lrow * 16 + lcol] = gx;
    red[256 + lrow * 16 + lcol] = gh;
    __syncthreads();
    if (lcol == 0) {
        float z = gb[0];
        for (int i = 0; i < 16; ++i)
            z += red[lrow * 16 + i] + red[256 + lrow * 16 + i];
        stat[lrow] = 1.f / (1.f + __expf(-z));
    }
    __syncthreads();
    float coeff = stat[lrow];
    for (int c = lcol; c < 512; c += 16) {
        float xv = x[grow * D_ + c];
        float hv = sC[lrow * 512 + c];
        out[grow * D_ + c] = coeff * xv + (1.f - coeff) * hv;
    }
}

// ---------------------------------------------------------------------------
// Launcher
// ---------------------------------------------------------------------------
extern "C" void kernel_launch(void* const* d_in, const int* in_sizes, int n_in,
                              void* d_out, int out_size, void* d_ws, size_t ws_size,
                              hipStream_t stream) {
    const float* x    = (const float*)d_in[0];
    const float* adj  = (const float*)d_in[1];
    const float* W_w  = (const float*)d_in[2];
    const float* W_b  = (const float*)d_in[3];
    const float* ln_g = (const float*)d_in[4];
    const float* ln_b = (const float*)d_in[5];
    const float* Wq   = (const float*)d_in[6];
    const float* bq   = (const float*)d_in[7];
    const float* Wk   = (const float*)d_in[8];
    const float* bk   = (const float*)d_in[9];
    const float* Wv   = (const float*)d_in[10];
    const float* bv   = (const float*)d_in[11];
    const float* Wo   = (const float*)d_in[12];
    const float* bo   = (const float*)d_in[13];
    const float* gw   = (const float*)d_in[14];
    const float* gb   = (const float*)d_in[15];
    const int*   use_adj = (const int*)d_in[16];
    float* out = (float*)d_out;

    const size_t SZ_ACT = (size_t)B_ * N_ * D_ * 2;   // 8 MB
    const size_t SZ_W   = (size_t)D_ * D_ * 2;        // 512 KB (packed)
    unsigned char* ws = (unsigned char*)d_ws;
    unsigned short* xbf  = (unsigned short*)(ws);
    unsigned short* wwpk = (unsigned short*)(ws + SZ_ACT);
    unsigned short* wqpk = (unsigned short*)(ws + SZ_ACT + 1 * SZ_W);
    unsigned short* wkpk = (unsigned short*)(ws + SZ_ACT + 2 * SZ_W);
    unsigned short* wvpk = (unsigned short*)(ws + SZ_ACT + 3 * SZ_W);
    unsigned short* wopk = (unsigned short*)(ws + SZ_ACT + 4 * SZ_W);
    unsigned short* hbf  = (unsigned short*)(ws + SZ_ACT + 5 * SZ_W);
    unsigned short* qbf  = (unsigned short*)(ws + 2 * SZ_ACT + 5 * SZ_W);
    unsigned short* kbf  = (unsigned short*)(ws + 3 * SZ_ACT + 5 * SZ_W);
    unsigned short* vpk  = (unsigned short*)(ws + 4 * SZ_ACT + 5 * SZ_W);
    unsigned short* ctxbf= (unsigned short*)(ws + 5 * SZ_ACT + 5 * SZ_W);

    const int nx = B_ * N_ * D_;
    k_cvt_bf16<<<(nx + 255) / 256, 256, 0, stream>>>(x, xbf, nx);
    k_pack_w<<<64, 256, 0, stream>>>(W_w, wwpk);
    k_pack_w<<<64, 256, 0, stream>>>(Wq,  wqpk);
    k_pack_w<<<64, 256, 0, stream>>>(Wk,  wkpk);
    k_pack_w<<<64, 256, 0, stream>>>(Wv,  wvpk);
    k_pack_w<<<64, 256, 0, stream>>>(Wo,  wopk);

    const int rowTiles = B_ * N_ / 16;  // 512
    k_gemm_ln<<<rowTiles, 256, GEMM_SMEM, stream>>>(xbf, wwpk, W_b, ln_g, ln_b, hbf);

    k_gemm_qkv<<<rowTiles, 256, QKV_SMEM, stream>>>(hbf, wqpk, bq, SCALE_, 0, qbf);
    k_gemm_qkv<<<rowTiles, 256, QKV_SMEM, stream>>>(hbf, wkpk, bk, 1.0f,   0, kbf);
    k_gemm_qkv<<<rowTiles, 256, QKV_SMEM, stream>>>(hbf, wvpk, bv, 1.0f,   1, vpk);

    k_attn<<<B_ * H_ * (N_ / 16), 256, ATT_SMEM, stream>>>(qbf, kbf, vpk, adj, use_adj, ctxbf);

    k_out<<<rowTiles, 256, GEMM_SMEM, stream>>>(ctxbf, wopk, bo, x, gw, gb, out);
}
